// TokenBottleneckAdapter_21595095564604
// MI455X (gfx1250) — matrix operand order repaired and checked by reading, not compiled, not run on hardware
//
#include <hip/hip_runtime.h>

typedef __attribute__((ext_vector_type(16))) _Float16 v16h;
typedef __attribute__((ext_vector_type(8)))  _Float16 v8h;
typedef __attribute__((ext_vector_type(4)))  _Float16 v4h;
typedef __attribute__((ext_vector_type(8)))  float    v8f;
typedef __attribute__((ext_vector_type(4)))  float    v4f;

#define C_DIM 1024
#define R_DIM 64
#define TOK_PER_WAVE 16
#define WAVES_PER_WG 4
#define TOK_PER_WG   (TOK_PER_WAVE * WAVES_PER_WG)
#define ZPITCH 1032   // halfs per z row (1024 + 8 pad -> conflict-free b128 A-frag loads)
#define HPITCH 72     // halfs per h row (64 + 8 pad)
#define OPITCH 132    // f32 per staged-output row (128 + 4 pad -> halfwave bank split)
#define WAVE_LDS_HALFS (TOK_PER_WAVE * ZPITCH + TOK_PER_WAVE * HPITCH)
#define TOTAL_TOK (16 * 4096)

// ---------------------------------------------------------------------------
// Branchless GELU (tanh form), ~6 VALU ops, no EXEC manipulation:
//   gelu(v) = 0.5 v (1 + tanh(c1 v + c2 v^3)) = v * (1 - 1/(exp(2(c1 v + c2 v^3)) + 1))
// v -> +inf: exp -> inf, rcp -> 0, result v.   v -> -inf: exp -> 0, result 0.
// ---------------------------------------------------------------------------
__device__ __forceinline__ float gelu_fast(float v) {
    float y = v * __builtin_fmaf(0.0713548162726f, v * v, 1.59576912161f);
    float e = __expf(y);                       // v_exp_f32
    float r = __builtin_amdgcn_rcpf(e + 1.0f); // v_rcp_f32
    return v - v * r;
}

// ---------------------------------------------------------------------------
// Prepass: convert both weight matrices to f16 in workspace (L2-resident).
// w_down is [R=64, C=1024] row-major  -> B[k][n] = wd[n][k], K contiguous.
// w_up   is [C=1024, R=64] row-major  -> B[k][n] = wu[n][k], K contiguous.
// ---------------------------------------------------------------------------
__global__ void cvt_weights_kernel(const float* __restrict__ wd,
                                   const float* __restrict__ wu,
                                   _Float16* __restrict__ wd16,
                                   _Float16* __restrict__ wu16) {
    int i = blockIdx.x * blockDim.x + threadIdx.x;
    if (i < R_DIM * C_DIM) {
        wd16[i] = (_Float16)wd[i];
        wu16[i] = (_Float16)wu[i];
    }
}

// ---------------------------------------------------------------------------
// Fused LN -> down-GEMM(WMMA f16) -> GELU -> up-GEMM(WMMA f16) -> residual.
// One wave owns 16 tokens; all LDS tiles are wave-private (no barriers).
// ---------------------------------------------------------------------------
__global__ void __launch_bounds__(WAVES_PER_WG * 32)
adapter_kernel(const float* __restrict__ x,
               const float* __restrict__ gamma,
               const float* __restrict__ beta,
               const _Float16* __restrict__ wd16,
               const _Float16* __restrict__ wu16,
               const float* __restrict__ scale_p,
               float* __restrict__ out) {
    extern __shared__ _Float16 lds[];
    const int lane = threadIdx.x & 31;
    const int wave = threadIdx.x >> 5;
    const int hi   = lane >> 4;      // half-wave select (0/1)
    const int rowm = lane & 15;      // M (A-frag) or N (B/D-frag) index within tile

    _Float16* zt = lds + wave * WAVE_LDS_HALFS;   // [16][ZPITCH] normalized f16
    _Float16* ht = zt + TOK_PER_WAVE * ZPITCH;    // [16][HPITCH] gelu(h) f16
    float*    ot = (float*)zt;                    // reuse z area: [16][OPITCH] f32
    const long tokBase = (long)blockIdx.x * TOK_PER_WG + (long)wave * TOK_PER_WAVE;

    // ---- preload gamma/beta (reused by all 16 rows) -------------------------
    v4f g[8], b[8];
#pragma unroll
    for (int c = 0; c < 8; ++c) {
        int col = 4 * lane + 128 * c;
        g[c] = *(const v4f*)(gamma + col);
        b[c] = *(const v4f*)(beta + col);
    }

    // ---- Phase 1: LayerNorm, row at a time; z (f16) -> LDS ------------------
    for (int r = 0; r < TOK_PER_WAVE; ++r) {
        const float* xr = x + (tokBase + r) * C_DIM;
        v4f xv[8];
        float s = 0.f, s2 = 0.f;
#pragma unroll
        for (int c = 0; c < 8; ++c) {
            xv[c] = *(const v4f*)(xr + 4 * lane + 128 * c);
#pragma unroll
            for (int e = 0; e < 4; ++e) { s += xv[c][e]; s2 += xv[c][e] * xv[c][e]; }
        }
#pragma unroll
        for (int m = 16; m >= 1; m >>= 1) {
            s  += __shfl_xor(s,  m, 32);
            s2 += __shfl_xor(s2, m, 32);
        }
        const float mu  = s * (1.0f / C_DIM);
        const float var = s2 * (1.0f / C_DIM) - mu * mu;
        const float rs  = rsqrtf(var + 1e-5f);
        _Float16* zr = zt + r * ZPITCH;
#pragma unroll
        for (int c = 0; c < 8; ++c) {
            v4h zv;
#pragma unroll
            for (int e = 0; e < 4; ++e)
                zv[e] = (_Float16)(((xv[c][e] - mu) * rs) * g[c][e] + b[c][e]);
            *(v4h*)(zr + 4 * lane + 128 * c) = zv;  // ds_store_b64, conflict-free
        }
    }

    // ---- Phase 2: down-projection h[16,64] = z[16,1024] @ WdT ---------------
    // A 16-bit 16x32 layout: lanes<16 K=k0+{0..7},{16..23}; lanes>=16 +8.
    // B 16-bit 32x16 layout: same K order as A: k0+hi*8+{0..7}, k0+16+hi*8+{0..7}.
    v8f acc[4] = {};
#pragma unroll 1
    for (int k0 = 0; k0 < C_DIM; k0 += 32) {
        v16h A;
        const _Float16* za = zt + rowm * ZPITCH + k0 + hi * 8;
        ((v8h*)&A)[0] = *(const v8h*)(za);        // K = k0 + hi*8 + 0..7
        ((v8h*)&A)[1] = *(const v8h*)(za + 16);   // K = k0 + 16 + hi*8 + 0..7
#pragma unroll
        for (int nt = 0; nt < 4; ++nt) {
            const _Float16* wb = wd16 + (nt * 16 + rowm) * C_DIM + k0 + hi * 8;
            v16h Bv;
            ((v8h*)&Bv)[0] = *(const v8h*)(wb);
            ((v8h*)&Bv)[1] = *(const v8h*)(wb + 16);
            acc[nt] = __builtin_amdgcn_wmma_f32_16x16x32_f16(
                false, A, false, Bv, (short)0, acc[nt], false, false);
        }
    }

    // ---- Phase 3: branchless GELU, repack D-layout -> LDS h tile (f16) ------
    // D layout: lane holds h[m = v + 8*hi][n = nt*16 + rowm] in acc[nt][v].
#pragma unroll
    for (int nt = 0; nt < 4; ++nt) {
#pragma unroll
        for (int v = 0; v < 8; ++v) {
            ht[(v + 8 * hi) * HPITCH + nt * 16 + rowm] = (_Float16)gelu_fast(acc[nt][v]);
        }
    }

    // ---- Phase 4: up-projection, staged epilogue with coalesced b128 I/O ----
    v16h A0, A1;
    {
        const _Float16* ha = ht + rowm * HPITCH + hi * 8;
        ((v8h*)&A0)[0] = *(const v8h*)(ha);        // K =  0 + hi*8 + 0..7
        ((v8h*)&A0)[1] = *(const v8h*)(ha + 16);   // K = 16 + hi*8 + 0..7
        ((v8h*)&A1)[0] = *(const v8h*)(ha + 32);   // K = 32 + hi*8 + 0..7
        ((v8h*)&A1)[1] = *(const v8h*)(ha + 48);   // K = 48 + hi*8 + 0..7
    }
    const float fscale = *scale_p;
#pragma unroll 1
    for (int grp = 0; grp < 8; ++grp) {          // 8 groups of 128 output columns
#pragma unroll
        for (int t = 0; t < 8; ++t) {            // 8 N-tiles of 16 per group
            const int nt = grp * 8 + t;
            const _Float16* wb = wu16 + (nt * 16 + rowm) * R_DIM;
            v16h B0, B1;
            ((v8h*)&B0)[0] = *(const v8h*)(wb + hi * 8);
            ((v8h*)&B0)[1] = *(const v8h*)(wb + hi * 8 + 16);
            ((v8h*)&B1)[0] = *(const v8h*)(wb + 32 + hi * 8);
            ((v8h*)&B1)[1] = *(const v8h*)(wb + 32 + hi * 8 + 16);
            v8f acu = {};
            acu = __builtin_amdgcn_wmma_f32_16x16x32_f16(
                false, A0, false, B0, (short)0, acu, false, false);
            acu = __builtin_amdgcn_wmma_f32_16x16x32_f16(
                false, A1, false, B1, (short)0, acu, false, false);
            // stage D tile in LDS (rows v+8*hi are 32 banks apart: conflict-free)
            const int col = t * 16 + rowm;
#pragma unroll
            for (int v = 0; v < 8; ++v)
                ot[(v + 8 * hi) * OPITCH + col] = acu[v];
        }
        // read back row-major; fully coalesced 512B/wave loads & stores
#pragma unroll 4
        for (int r = 0; r < TOK_PER_WAVE; ++r) {
            const long idx = (tokBase + r) * C_DIM + grp * 128 + lane * 4;
            v4f xv = *(const v4f*)(x + idx);
            v4f ov = *(const v4f*)(ot + r * OPITCH + lane * 4);
            v4f res;
#pragma unroll
            for (int e = 0; e < 4; ++e) res[e] = xv[e] + fscale * ov[e];
            *(v4f*)(out + idx) = res;
        }
    }
}

// ---------------------------------------------------------------------------
extern "C" void kernel_launch(void* const* d_in, const int* in_sizes, int n_in,
                              void* d_out, int out_size, void* d_ws, size_t ws_size,
                              hipStream_t stream) {
    const float* x       = (const float*)d_in[0];
    const float* gamma   = (const float*)d_in[1];
    const float* beta    = (const float*)d_in[2];
    const float* w_down  = (const float*)d_in[3];   // [64, 1024]
    const float* w_up    = (const float*)d_in[4];   // [1024, 64]
    const float* scale_p = (const float*)d_in[5];
    float* out = (float*)d_out;

    _Float16* wd16 = (_Float16*)d_ws;
    _Float16* wu16 = wd16 + R_DIM * C_DIM;

    cvt_weights_kernel<<<(R_DIM * C_DIM + 255) / 256, 256, 0, stream>>>(
        w_down, w_up, wd16, wu16);

    const size_t lds_bytes = (size_t)WAVES_PER_WG * WAVE_LDS_HALFS * sizeof(_Float16);
    adapter_kernel<<<TOTAL_TOK / TOK_PER_WG, WAVES_PER_WG * 32, lds_bytes, stream>>>(
        x, gamma, beta, wd16, wu16, scale_p, out);
}
